// LSTMAutoencoder_23785528885371
// MI455X (gfx1250) — compile-verified, matrix operand order checked by reading, not run on hardware
//
#include <hip/hip_runtime.h>

// ---------------------------------------------------------------------------
// LSTM autoencoder on gfx1250 (MI455X): persistent-scan bf16 WMMA kernels.
// B=2048, T=100, In=64, H=128, Lat=32, G=4H=512.
// Batch rows are independent in an LSTM recurrence, so each 16-row block runs
// entire time scans privately: h is carried in LDS (bf16, already in A-matrix
// staging layout), c is carried in registers. Only 2 compute kernels total.
// ---------------------------------------------------------------------------

typedef __bf16 bf16_t;
typedef __attribute__((ext_vector_type(16))) __bf16 v16bf;
typedef __attribute__((ext_vector_type(8)))  __bf16 v8bf;
typedef __attribute__((ext_vector_type(8)))  float  v8f;
typedef int v4i_ __attribute__((vector_size(16)));

#define B_  2048
#define T_  100
#define IN_ 64
#define H_  128
#define G_  512

#define AS1 __attribute__((address_space(1)))
#define AS3 __attribute__((address_space(3)))

#if __has_builtin(__builtin_amdgcn_global_load_async_to_lds_b128)
#define HAVE_ASYNC_LDS 1
#else
#define HAVE_ASYNC_LDS 0
#endif

__device__ __forceinline__ void async_wait0() {
#if __has_builtin(__builtin_amdgcn_s_wait_asynccnt)
  __builtin_amdgcn_s_wait_asynccnt(0);
#else
  asm volatile("s_wait_asynccnt 0x0" ::: "memory");
#endif
}

__device__ __forceinline__ float sigf(float x) { return 1.0f / (1.0f + __expf(-x)); }

__device__ __forceinline__ v16bf cat8(v8bf lo, v8bf hi) {
  return __builtin_shufflevector(lo, hi, 0,1,2,3,4,5,6,7,8,9,10,11,12,13,14,15);
}

// Launder a uniform value through an SGPR so the optimizer cannot constant-
// propagate it: keeps K-loops rolled so weight-fragment loads stream from L2
// inside the loop instead of being hoisted out of the t-loop and spilled.
__device__ __forceinline__ int opaque_s(int v) {
  asm volatile("" : "+s"(v));
  return v;
}

// ---------------------------------------------------------------------------
// Build WMMA-fragment-ready bf16 weights.
// src: N x K row-major f32 (W[n][k]); B-matrix element (k,n) = src[n*K+k].
// dst: for k-block kb=k/32: element ((kb*N + n)*32 + pos(k%32)) where pos
// reorders K to {0..7,16..23,8..15,24..31} so lo/hi half-lanes each read two
// contiguous 16B chunks (matches 16-bit WMMA B fragment layout).
// ---------------------------------------------------------------------------
__global__ void build_wfrag_kernel(const float* __restrict__ src, bf16_t* __restrict__ dst,
                                   int N, int K, int kofs) {
  int idx = blockIdx.x * 256 + threadIdx.x;
  if (idx >= N * K) return;
  int n = idx / K, k = idx - n * K;
  int kk = k + kofs;
  int kb = kk >> 5, kr = kk & 31;
  int pos = (kr >= 8 && kr < 24) ? (kr ^ 24) : kr;
  dst[((size_t)kb * N + n) * 32 + pos] = (bf16_t)src[(size_t)n * K + k];
}

// ---------------------------------------------------------------------------
// 16 x 512 GEMM tile: Gt(16x512) = A(16xKtot, bf16 LDS) @ W(Ktot x 512 frag).
// 8 waves, each owns 64 gate columns (4 wmma n-tiles). No internal barriers.
// ---------------------------------------------------------------------------
__device__ __forceinline__ void gemm_to_gt(const bf16_t* __restrict__ A, int Kpad, int Ktot,
                                           const bf16_t* __restrict__ W,
                                           float* __restrict__ Gt, int lane, int wave) {
  const int l16 = lane & 15;
  const int khi = (lane & 16) >> 1;   // lo lanes K {0..7,16..23}, hi lanes +8
  const int ng  = wave;
  v8f acc[4];
#pragma unroll
  for (int i = 0; i < 4; ++i)
#pragma unroll
    for (int e = 0; e < 8; ++e) acc[i][e] = 0.0f;

  const int kt = opaque_s(Ktot);      // keep loop rolled: stream W from L2
  for (int kb = 0; kb < kt; kb += 32) {
    const v8bf* ap = (const v8bf*)&A[l16 * Kpad + kb + khi];
    v16bf a = cat8(ap[0], ap[2]);
#pragma unroll
    for (int nt = 0; nt < 4; ++nt) {
      int nb = ng * 64 + nt * 16;
      const v8bf* bp = (const v8bf*)&W[(((size_t)(kb >> 5) * G_) + nb + l16) * 32 + (lane & 16)];
      v16bf bm = cat8(bp[0], bp[1]);
      acc[nt] = __builtin_amdgcn_wmma_f32_16x16x32_bf16(false, a, false, bm,
                                                        (short)0, acc[nt], false, false);
    }
    // speculative prefetch of next weight k-slice
    __builtin_prefetch(&W[(((size_t)((kb >> 5) + 1) * G_) + ng * 64 + l16) * 32], 0, 3);
  }
#pragma unroll
  for (int nt = 0; nt < 4; ++nt) {
    int nb = ng * 64 + nt * 16;
#pragma unroll
    for (int r = 0; r < 8; ++r)
      Gt[(((lane >> 4) << 3) + r) * G_ + nb + l16] = acc[nt][r];
  }
}

// ---------------------------------------------------------------------------
// Encoder: 2-layer biLSTM + fc_z + LayerNorm, one block per 16 batch rows.
// LDS: Abuf 16x392 bf16 @0 | Gt 16x512 f32 @12544 | last 16x256 f32 @45312.
// ---------------------------------------------------------------------------
__global__ void encoder_kernel(
    const float* __restrict__ x,
    const bf16_t* __restrict__ We0f, const bf16_t* __restrict__ We0b,
    const bf16_t* __restrict__ We1f, const bf16_t* __restrict__ We1b,
    const float* __restrict__ be0f, const float* __restrict__ be0b,
    const float* __restrict__ be1f, const float* __restrict__ be1b,
    bf16_t* __restrict__ h0cat,               // (T,B,256) bf16
    const float* __restrict__ fczW, const float* __restrict__ fczb,
    const float* __restrict__ lng, const float* __restrict__ lnb,
    float* __restrict__ zout)                 // (B,32)
{
  extern __shared__ char smem[];
  bf16_t* Abuf = (bf16_t*)smem;
  float*  Gt   = (float*)(smem + 12544);
  float*  last = (float*)(smem + 12544 + 32768);
  const int tid  = threadIdx.x;
  const int m0   = blockIdx.x * 16;
  const int wave = tid >> 5, lane = tid & 31;
  const int KP0 = 200, KP1 = 392;             // 192+8, 384+8 (16B-aligned rows)
  float c[8];

  // ===================== layer 0, both directions =====================
#pragma unroll
  for (int dir = 0; dir < 2; ++dir) {
    const bf16_t* W    = dir ? We0b : We0f;
    const float*  bias = dir ? be0b : be0f;
    for (int i = tid; i < 16 * 128; i += 256)
      Abuf[(i >> 7) * KP0 + 64 + (i & 127)] = (bf16_t)0.0f;
#pragma unroll
    for (int s = 0; s < 8; ++s) c[s] = 0.0f;
    __syncthreads();
    for (int it = 0; it < T_; ++it) {
      int t = dir ? (T_ - 1 - it) : it;
      for (int i = tid; i < 16 * 64; i += 256) {
        int r = i >> 6, k = i & 63;
        Abuf[r * KP0 + k] = (bf16_t)x[(size_t)(m0 + r) * T_ * IN_ + (size_t)t * IN_ + k];
      }
      __syncthreads();
      gemm_to_gt(Abuf, KP0, 192, W, Gt, lane, wave);
      __syncthreads();
#pragma unroll
      for (int s = 0; s < 8; ++s) {
        int idx = tid + s * 256, r = idx >> 7, j = idx & 127;
        float gi = Gt[r * G_ + j]       + bias[j];
        float gf = Gt[r * G_ + 128 + j] + bias[128 + j];
        float gg = Gt[r * G_ + 256 + j] + bias[256 + j];
        float go = Gt[r * G_ + 384 + j] + bias[384 + j];
        float cn = sigf(gf) * c[s] + sigf(gi) * tanhf(gg);
        float hn = sigf(go) * tanhf(cn);
        c[s] = cn;
        bf16_t hb = (bf16_t)hn;
        Abuf[r * KP0 + 64 + j] = hb;                               // recurrent feed
        h0cat[((size_t)t * B_ + m0 + r) * 256 + dir * 128 + j] = hb;
      }
      __syncthreads();
    }
  }

  __threadfence();      // our own h0cat stores -> visible to our own loads
  __syncthreads();

  // ===================== layer 1 forward (keep only final h) ==========
  for (int i = tid; i < 16 * 128; i += 256)
    Abuf[(i >> 7) * KP1 + 256 + (i & 127)] = (bf16_t)0.0f;
#pragma unroll
  for (int s = 0; s < 8; ++s) c[s] = 0.0f;
  __syncthreads();
  for (int t = 0; t < T_; ++t) {
    // stage h0cat tile (16 rows x 256 bf16) into Abuf x-region, 16B chunks
    for (int i = tid; i < 512; i += 256) {
      int r = i >> 5, ch = i & 31;
      const bf16_t* gsrc = &h0cat[((size_t)t * B_ + m0 + r) * 256 + ch * 8];
      bf16_t* ldst = &Abuf[r * KP1 + ch * 8];
#if HAVE_ASYNC_LDS
      __builtin_amdgcn_global_load_async_to_lds_b128(
          (AS1 v4i_*)(unsigned long long)gsrc,
          (AS3 v4i_*)(unsigned)((char*)ldst - smem), 0, 0);
#else
      *(v8bf*)ldst = *(const v8bf*)gsrc;
#endif
    }
#if HAVE_ASYNC_LDS
    async_wait0();
#endif
    __syncthreads();
    gemm_to_gt(Abuf, KP1, 384, We1f, Gt, lane, wave);
    __syncthreads();
#pragma unroll
    for (int s = 0; s < 8; ++s) {
      int idx = tid + s * 256, r = idx >> 7, j = idx & 127;
      float gi = Gt[r * G_ + j]       + be1f[j];
      float gf = Gt[r * G_ + 128 + j] + be1f[128 + j];
      float gg = Gt[r * G_ + 256 + j] + be1f[256 + j];
      float go = Gt[r * G_ + 384 + j] + be1f[384 + j];
      float cn = sigf(gf) * c[s] + sigf(gi) * tanhf(gg);
      float hn = sigf(go) * tanhf(cn);
      c[s] = cn;
      Abuf[r * KP1 + 256 + j] = (bf16_t)hn;
      if (t == T_ - 1) last[r * 256 + j] = hn;
    }
    __syncthreads();
  }

  // ===================== layer 1 backward: only first reverse step ====
  for (int i = tid; i < 16 * 128; i += 256)
    Abuf[(i >> 7) * KP1 + 256 + (i & 127)] = (bf16_t)0.0f;
  for (int i = tid; i < 512; i += 256) {
    int r = i >> 5, ch = i & 31;
    const bf16_t* gsrc = &h0cat[((size_t)(T_ - 1) * B_ + m0 + r) * 256 + ch * 8];
    bf16_t* ldst = &Abuf[r * KP1 + ch * 8];
#if HAVE_ASYNC_LDS
    __builtin_amdgcn_global_load_async_to_lds_b128(
        (AS1 v4i_*)(unsigned long long)gsrc,
        (AS3 v4i_*)(unsigned)((char*)ldst - smem), 0, 0);
#else
    *(v8bf*)ldst = *(const v8bf*)gsrc;
#endif
  }
#if HAVE_ASYNC_LDS
  async_wait0();
#endif
  __syncthreads();
  gemm_to_gt(Abuf, KP1, 384, We1b, Gt, lane, wave);
  __syncthreads();
#pragma unroll
  for (int s = 0; s < 8; ++s) {
    int idx = tid + s * 256, r = idx >> 7, j = idx & 127;
    float gi = Gt[r * G_ + j]       + be1b[j];
    float gg = Gt[r * G_ + 256 + j] + be1b[256 + j];
    float go = Gt[r * G_ + 384 + j] + be1b[384 + j];
    float cn = sigf(gi) * tanhf(gg);           // c starts at 0 on first step
    float hn = sigf(go) * tanhf(cn);
    last[r * 256 + 128 + j] = hn;
  }
  __syncthreads();

  // ===================== fc_z + LayerNorm (per-row, Lat=32) ===========
  for (int o = tid; o < 512; o += 256) {       // zz into Gt[0..511]
    int r = o >> 5, cc = o & 31;
    float acc = fczb[cc];
    for (int k = 0; k < 256; ++k) acc += last[r * 256 + k] * fczW[cc * 256 + k];
    Gt[o] = acc;
  }
  __syncthreads();
  if (tid < 16) {
    float s = 0.0f, q = 0.0f;
    for (int cc = 0; cc < 32; ++cc) s += Gt[tid * 32 + cc];
    float mu = s * (1.0f / 32.0f);
    for (int cc = 0; cc < 32; ++cc) { float d = Gt[tid * 32 + cc] - mu; q += d * d; }
    Gt[512 + tid] = mu;
    Gt[528 + tid] = rsqrtf(q * (1.0f / 32.0f) + 1e-5f);
  }
  __syncthreads();
  for (int o = tid; o < 512; o += 256) {
    int r = o >> 5, cc = o & 31;
    zout[(size_t)(m0 + r) * 32 + cc] = (Gt[o] - Gt[512 + r]) * Gt[528 + r] * lng[cc] + lnb[cc];
  }
}

// ---------------------------------------------------------------------------
// Decoder: 2-layer autoregressive LSTM + fused fc_out, full scan per block.
// h_init computed in-block from z (raw (B,2H)->(2,B,H) reshape only *reads*
// other rows of z, which the previous kernel produced).
// LDS: A0 16x200 bf16 @0 | A1 16x264 bf16 @6400 | Gt 16x512 f32 @14848.
// ---------------------------------------------------------------------------
__global__ void decoder_kernel(
    const float* __restrict__ z,               // (B,32)
    const float* __restrict__ fiW, const float* __restrict__ fib,  // fc_init
    const bf16_t* __restrict__ Wd0, const float* __restrict__ bd0,
    const bf16_t* __restrict__ Wd1, const float* __restrict__ bd1,
    const bf16_t* __restrict__ fcW, const float* __restrict__ fcb,
    float* __restrict__ xhat)                  // (B,T,64)
{
  extern __shared__ char smem[];
  bf16_t* A0 = (bf16_t*)smem;
  bf16_t* A1 = (bf16_t*)(smem + 6400);
  float*  Gt = (float*)(smem + 14848);
  const int tid  = threadIdx.x;
  const int m0   = blockIdx.x * 16;
  const int wave = tid >> 5, lane = tid & 31;
  const int l16  = lane & 15;
  const int khi  = (lane & 16) >> 1;
  const int KP0 = 200, KP1 = 264;
  float c0[8], c1[8];

  // init: x region zero, h regions from fc_init(z) with raw reshape, c = 0
  for (int i = tid; i < 16 * 64; i += 256) A0[(i >> 6) * KP0 + (i & 63)] = (bf16_t)0.0f;
  for (int i = tid; i < 16 * 128; i += 256) {
    int r = i >> 7, j = i & 127;
    int flat = (m0 + r) * H_ + j;              // layer 0: l = 0
    {
      int sr = flat >> 8, sc = flat & 255;
      float acc = fib[sc];
      for (int k = 0; k < 32; ++k) acc += z[(size_t)sr * 32 + k] * fiW[sc * 32 + k];
      A0[r * KP0 + 64 + j] = (bf16_t)acc;
    }
    {
      int flat1 = B_ * H_ + flat;              // layer 1: l = 1
      int sr = flat1 >> 8, sc = flat1 & 255;
      float acc = fib[sc];
      for (int k = 0; k < 32; ++k) acc += z[(size_t)sr * 32 + k] * fiW[sc * 32 + k];
      A1[r * KP1 + 128 + j] = (bf16_t)acc;
    }
  }
#pragma unroll
  for (int s = 0; s < 8; ++s) { c0[s] = 0.0f; c1[s] = 0.0f; }
  __syncthreads();

  for (int t = 0; t < T_; ++t) {
    // ---- dec layer 0 ----
    gemm_to_gt(A0, KP0, 192, Wd0, Gt, lane, wave);
    __syncthreads();
#pragma unroll
    for (int s = 0; s < 8; ++s) {
      int idx = tid + s * 256, r = idx >> 7, j = idx & 127;
      float gi = Gt[r * G_ + j]       + bd0[j];
      float gf = Gt[r * G_ + 128 + j] + bd0[128 + j];
      float gg = Gt[r * G_ + 256 + j] + bd0[256 + j];
      float go = Gt[r * G_ + 384 + j] + bd0[384 + j];
      float cn = sigf(gf) * c0[s] + sigf(gi) * tanhf(gg);
      float hn = sigf(go) * tanhf(cn);
      c0[s] = cn;
      bf16_t hb = (bf16_t)hn;
      A0[r * KP0 + 64 + j] = hb;     // dec0 recurrent feed
      A1[r * KP1 + j]      = hb;     // dec1 input
    }
    __syncthreads();
    // ---- dec layer 1 ----
    gemm_to_gt(A1, KP1, 256, Wd1, Gt, lane, wave);
    __syncthreads();
#pragma unroll
    for (int s = 0; s < 8; ++s) {
      int idx = tid + s * 256, r = idx >> 7, j = idx & 127;
      float gi = Gt[r * G_ + j]       + bd1[j];
      float gf = Gt[r * G_ + 128 + j] + bd1[128 + j];
      float gg = Gt[r * G_ + 256 + j] + bd1[256 + j];
      float go = Gt[r * G_ + 384 + j] + bd1[384 + j];
      float cn = sigf(gf) * c1[s] + sigf(gi) * tanhf(gg);
      float hn = sigf(go) * tanhf(cn);
      c1[s] = cn;
      A1[r * KP1 + 128 + j] = (bf16_t)hn;   // dec1 recurrent feed (also fc input)
    }
    __syncthreads();
    // ---- fused fc_out: pred(16x64) = h1(16x128) @ fcW ----
    if (wave < 4) {
      v8f pa;
#pragma unroll
      for (int e = 0; e < 8; ++e) pa[e] = 0.0f;
      const int kend = opaque_s(H_);
      for (int kb = 0; kb < kend; kb += 32) {
        const v8bf* ap = (const v8bf*)&A1[l16 * KP1 + 128 + kb + khi];
        v16bf a = cat8(ap[0], ap[2]);
        const v8bf* bp = (const v8bf*)&fcW[(((size_t)(kb >> 5) * 64) + wave * 16 + l16) * 32 + (lane & 16)];
        v16bf bm = cat8(bp[0], bp[1]);
        pa = __builtin_amdgcn_wmma_f32_16x16x32_bf16(false, a, false, bm,
                                                     (short)0, pa, false, false);
      }
#pragma unroll
      for (int r = 0; r < 8; ++r) {
        int row = ((lane >> 4) << 3) + r;
        int col = wave * 16 + l16;
        float pv = pa[r] + fcb[col];
        xhat[(size_t)(m0 + row) * T_ * IN_ + (size_t)t * IN_ + col] = pv;
        A0[row * KP0 + col] = (bf16_t)pv;    // autoregressive feedback
      }
    }
    __syncthreads();
  }
}

// ---------------------------------------------------------------------------
extern "C" void kernel_launch(void* const* d_in, const int* in_sizes, int n_in,
                              void* d_out, int out_size, void* d_ws, size_t ws_size,
                              hipStream_t stream) {
  const float* x         = (const float*)d_in[0];
  const float* enc0_Wih  = (const float*)d_in[1];
  const float* enc0_Whh  = (const float*)d_in[2];
  const float* enc0_b    = (const float*)d_in[3];
  const float* enc1_Wih  = (const float*)d_in[4];
  const float* enc1_Whh  = (const float*)d_in[5];
  const float* enc1_b    = (const float*)d_in[6];
  const float* fc_z_W    = (const float*)d_in[7];
  const float* fc_z_b    = (const float*)d_in[8];
  const float* ln_g      = (const float*)d_in[9];
  const float* ln_b      = (const float*)d_in[10];
  const float* fc_init_W = (const float*)d_in[11];
  const float* fc_init_b = (const float*)d_in[12];
  const float* dec0_Wih  = (const float*)d_in[13];
  const float* dec0_Whh  = (const float*)d_in[14];
  const float* dec0_b    = (const float*)d_in[15];
  const float* dec1_Wih  = (const float*)d_in[16];
  const float* dec1_Whh  = (const float*)d_in[17];
  const float* dec1_b    = (const float*)d_in[18];
  const float* fc_out_W  = (const float*)d_in[19];
  const float* fc_out_b  = (const float*)d_in[20];
  (void)in_sizes; (void)n_in; (void)out_size; (void)ws_size;

  char* ws = (char*)d_ws;
  size_t off = 0;
  auto alloc = [&](size_t bytes) -> void* {
    void* p = ws + off;
    off = (off + bytes + 255) & ~(size_t)255;
    return p;
  };
  bf16_t* Wc_e0f = (bf16_t*)alloc((size_t)192 * 512 * 2);
  bf16_t* Wc_e0b = (bf16_t*)alloc((size_t)192 * 512 * 2);
  bf16_t* Wc_e1f = (bf16_t*)alloc((size_t)384 * 512 * 2);
  bf16_t* Wc_e1b = (bf16_t*)alloc((size_t)384 * 512 * 2);
  bf16_t* Wc_d0  = (bf16_t*)alloc((size_t)192 * 512 * 2);
  bf16_t* Wc_d1  = (bf16_t*)alloc((size_t)256 * 512 * 2);
  bf16_t* fcWb   = (bf16_t*)alloc((size_t)128 * 64 * 2);
  bf16_t* h0cat  = (bf16_t*)alloc((size_t)T_ * B_ * 256 * 2);   // layer0 ys, bf16

  auto bw = [&](const float* s, bf16_t* d, int N, int K, int kofs) {
    build_wfrag_kernel<<<(N * K + 255) / 256, 256, 0, stream>>>(s, d, N, K, kofs);
  };
  bw(enc0_Wih,              Wc_e0f, 512,  64,   0);
  bw(enc0_Whh,              Wc_e0f, 512, 128,  64);
  bw(enc0_Wih + 512 * 64,   Wc_e0b, 512,  64,   0);
  bw(enc0_Whh + 512 * 128,  Wc_e0b, 512, 128,  64);
  bw(enc1_Wih,              Wc_e1f, 512, 256,   0);
  bw(enc1_Whh,              Wc_e1f, 512, 128, 256);
  bw(enc1_Wih + 512 * 256,  Wc_e1b, 512, 256,   0);
  bw(enc1_Whh + 512 * 128,  Wc_e1b, 512, 128, 256);
  bw(dec0_Wih,              Wc_d0,  512,  64,   0);
  bw(dec0_Whh,              Wc_d0,  512, 128,  64);
  bw(dec1_Wih,              Wc_d1,  512, 128,   0);
  bw(dec1_Whh,              Wc_d1,  512, 128, 128);
  bw(fc_out_W,              fcWb,    64, 128,   0);

  float* xhat = (float*)d_out;
  float* zptr = xhat + (size_t)B_ * T_ * IN_;

  encoder_kernel<<<B_ / 16, 256, 12544 + 32768 + 16384, stream>>>(
      x, Wc_e0f, Wc_e0b, Wc_e1f, Wc_e1b,
      enc0_b, enc0_b + 512, enc1_b, enc1_b + 512,
      h0cat, fc_z_W, fc_z_b, ln_g, ln_b, zptr);

  decoder_kernel<<<B_ / 16, 256, 6400 + 8448 + 32768, stream>>>(
      zptr, fc_init_W, fc_init_b,
      Wc_d0, dec0_b, Wc_d1, dec1_b, fcWb, fc_out_b, xhat);
}
